// HumanBrain_15195594293747
// MI455X (gfx1250) — compile-verified, matrix operand order checked by reading, not run on hardware
//
#include <hip/hip_runtime.h>
#include <hip/hip_bf16.h>
#include <cstdint>
#include <cstddef>

// ---------------------------------------------------------------------------
// HumanBrain forward on MI455X (gfx1250, wave32, WMMA bf16 16x16x32).
// Persistent batch-tiled recurrence kernel + WMMA epilogue kernels.
// A-operands are kept in bf16 LDS mirrors so the WMMA inner loops carry no
// conversion VALU (ds_load_b128 -> v_wmma, co-executing with elementwise VALU).
// ---------------------------------------------------------------------------

typedef __attribute__((ext_vector_type(16))) __bf16 v16bf;
typedef __attribute__((ext_vector_type(8)))  float  v8f;

#define NTHR   512
#define NWAVES 16

// ----- model dimensions ----------------------------------------------------
constexpr int Bm_ = 128, Tm_ = 256, Hm_ = 512, IN_ = 32;

constexpr int   RH_[13]   = {25,40,30,51,30,51,40,71,71,25,30,20,20};
constexpr int   CO_[13]   = {0,25,65,95,146,176,227,267,338,409,434,464,484};
constexpr int   RKT_[13]  = {3,3,3,4,4,6,6,9,9,6,7,4,7};
constexpr int   RNT_[13]  = {2,3,2,4,2,4,3,5,5,2,2,2,2};
constexpr int   ROFF_[13] = {0,3072,7680,10752,18944,23040,35328,44544,67584,
                             90624,96768,103936,108032};
constexpr float RDTS_[13] = {3.0f,2.5f,2.0f,1.5f,2.0f,0.8f,1.8f,0.25f,0.3f,
                             1.2f,0.6f,0.5f,1.5f};

// bf16 weight arena offsets (elements)
constexpr int ATH1  = 115200;     // thalamus_lin  (K=512pad, N=512) 16x32 tiles
constexpr int ATH2  = 377344;     // thalamic_proj
constexpr int ANEXT = 639488;     // next_lin (K=512, N=32)

// LDS geometry
constexpr int CBW = 504;  // f32 state (13 regions concatenated, real widths)
constexpr int ABW = 512;  // bf16 state mirror, K-padded for thalamus tiles
constexpr int XS  = 288;  // bf16 region-input concat (max padded K = 288)
constexpr int TGW = 72;   // target staging width (max h = 71)

// ----- output layout (floats, concatenated return order) -------------------
constexpr size_t OUT_PRED  = 0;          // [128,255,32]
constexpr size_t OUT_HID   = 1044480;    // [128,256,512]
constexpr size_t OUT_ANOM  = 17821696;   // [128,256]
constexpr size_t OUT_SUR   = 17854464;   // [128,255]
constexpr size_t OUT_PROPS = 17887104;   // [128,8]
constexpr long long TROFF_[13] = {       // traces, [128,256,h_r] each
  17888128LL,18707328LL,20018048LL,21001088LL,22672256LL,23655296LL,
  25326464LL,26637184LL,28963712LL,31290240LL,32109440LL,33092480LL,33747840LL};

// ----- workspace layout (floats) -------------------------------------------
constexpr size_t WS_THAL = 327936;                 // after bf16 arena
constexpr size_t WS_PROP = WS_THAL + 8ull*16*512;  // props scratch

struct Ptrs { const float* q[128]; };

// ----- helpers --------------------------------------------------------------
__device__ __forceinline__ __bf16 f2bf(float f) {
  unsigned u = __builtin_bit_cast(unsigned, f);
  unsigned short s = (unsigned short)((u + 0x7FFFu + ((u >> 16) & 1u)) >> 16);
  __bf16 r;
  __builtin_memcpy(&r, &s, 2);
  return r;
}

__device__ __forceinline__ float bf2f(__bf16 b) {
  unsigned short s;
  __builtin_memcpy(&s, &b, 2);
  unsigned u = ((unsigned)s) << 16;
  return __builtin_bit_cast(float, u);
}

__device__ __forceinline__ v8f wmma_bf16(v16bf a, v16bf b, v8f c) {
#if defined(__gfx1250__)
  return __builtin_amdgcn_wmma_f32_16x16x32_bf16(false, a, false, b,
                                                 (short)0, c, false, false);
#else
  return c;
#endif
}

// C tile = sum_kt A(16x32 bf16, LDS rows) x B(32x16 bf16, packed arena tiles)
__device__ __forceinline__ v8f gemm_bf16A(const __bf16* Ab, int ldA, int kT,
                                          const __bf16* wt, int nt, int lane) {
  v8f acc = {0.f,0.f,0.f,0.f,0.f,0.f,0.f,0.f};
  const int m  = lane & 15;
  const int hf = lane >> 4;
  const __bf16* srow  = Ab + m * ldA + hf * 16;
  const __bf16* wbase = wt + (((size_t)nt * kT) << 9) + (lane << 4);
  for (int kt = 0; kt < kT; ++kt) {
    v16bf a = *(const v16bf*)(srow + kt * 32);
    v16bf b = *(const v16bf*)(wbase + ((size_t)kt << 9));
    if (kt + 1 < kT)
      __builtin_prefetch((const void*)(wbase + (((size_t)kt + 1) << 9)), 0, 1);
    acc = wmma_bf16(a, b, acc);
  }
  return acc;
}

__device__ __forceinline__ void seg(__bf16 (*xin)[XS], const float (*cb)[CBW],
                                    int dcol, int scol, int len, int tid) {
  for (int e = tid; e < 16 * len; e += NTHR) {
    int row = e / len, c = e - row * len;
    xin[row][dcol + c] = f2bf(cb[row][scol + c]);
  }
}

__device__ __forceinline__ void seg_inp(__bf16 (*xin)[XS], const float (*ib)[40],
                                        int len, int tid) {
  for (int e = tid; e < 16 * len; e += NTHR) {
    int row = e / len, c = e - row * len;
    xin[row][c] = f2bf(ib[row][c]);
  }
}

// xin[row][dcol+o] += scale * sigmoid(gate) * tanh(proj), input from f32 state
__device__ __forceinline__ void axon_add_x(__bf16 (*xin)[XS], int dcol, int to,
                                           const float (*cb)[CBW], int scol, int fr,
                                           const float* pw, const float* pb,
                                           const float* gw, const float* gb,
                                           float scale, int tid) {
  for (int e = tid; e < 16 * to; e += NTHR) {
    int row = e / to, o = e - row * to;
    float gp = gb[o], pp = pb[o];
    for (int k = 0; k < fr; ++k) {
      float v = cb[row][scol + k];
      gp += gw[(size_t)o * fr + k] * v;
      pp += pw[(size_t)o * fr + k] * v;
    }
    float add = scale * (1.f / (1.f + __expf(-gp))) * tanhf(pp);
    xin[row][dcol + o] = f2bf(bf2f(xin[row][dcol + o]) + add);
  }
}

// state[row][dcol+o] += scale*axon (post-region additions), keep bf16 mirror
__device__ __forceinline__ void axon_add_cb(float (*cb)[CBW], __bf16 (*ab)[ABW],
                                            int dcol, int to, int scol, int fr,
                                            const float* pw, const float* pb,
                                            const float* gw, const float* gb,
                                            float scale, int tid) {
  for (int e = tid; e < 16 * to; e += NTHR) {
    int row = e / to, o = e - row * to;
    float gp = gb[o], pp = pb[o];
    for (int k = 0; k < fr; ++k) {
      float v = cb[row][scol + k];
      gp += gw[(size_t)o * fr + k] * v;
      pp += pw[(size_t)o * fr + k] * v;
    }
    float nv = cb[row][dcol + o] +
               scale * (1.f / (1.f + __expf(-gp))) * tanhf(pp);
    cb[row][dcol + o] = nv;
    ab[row][dcol + o] = f2bf(nv);
  }
}

// GEMM -> tanh target -> liquid decay update -> LayerNorm -> state (f32+bf16)
__device__ __forceinline__ void region_stage(int r, const Ptrs& p,
    float (*cb)[CBW], __bf16 (*ab)[ABW], __bf16 (*xinb)[XS],
    float (*tgt)[TGW], float (*mv)[2],
    const __bf16* arena, int tid, int lane, int wave) {
  const int h = RH_[r], co = CO_[r], kT = RKT_[r], nT = RNT_[r];
  const float* wb  = p.q[2 + 6*r];  // Win.b
  const float* lsp = p.q[4 + 6*r];  // log_step
  const float* gm  = p.q[5 + 6*r];  // gamma
  const float* bt  = p.q[6 + 6*r];  // beta
  const __bf16* wt = arena + ROFF_[r];

  for (int nt = wave; nt < nT; nt += NWAVES) {
    v8f acc = gemm_bf16A(&xinb[0][0], XS, kT, wt, nt, lane);
    int col   = nt * 16 + (lane & 15);
    int rbase = (lane >> 4) * 8;
    if (col < h) {
      float bia = wb[col];
#pragma unroll
      for (int i = 0; i < 8; ++i) tgt[rbase + i][col] = tanhf(acc[i] + bia);
    }
  }
  __syncthreads();
  const float dts = RDTS_[r];
  for (int e = tid; e < 16 * h; e += NTHR) {
    int row = e / h, c = e - row * h;
    float dec = __expf(-dts * __expf(lsp[c]));          // DT = 1.0
    tgt[row][c] = cb[row][co + c] * dec + (1.f - dec) * tgt[row][c];
  }
  __syncthreads();
  if (tid < 16) {
    float mu = 0.f;
    for (int c = 0; c < h; ++c) mu += tgt[tid][c];
    mu /= (float)h;
    float va = 0.f;
    for (int c = 0; c < h; ++c) { float d = tgt[tid][c] - mu; va += d * d; }
    va /= (float)h;
    mv[tid][0] = mu;
    mv[tid][1] = rsqrtf(va + 1e-5f);
  }
  __syncthreads();
  for (int e = tid; e < 16 * h; e += NTHR) {
    int row = e / h, c = e - row * h;
    float nv = (tgt[row][c] - mv[row][0]) * mv[row][1] * gm[c] + bt[c];
    cb[row][co + c] = nv;
    ab[row][co + c] = f2bf(nv);
  }
  __syncthreads();
}

// ----- weight packing: fp32 [Win|Wh] -> bf16 WMMA-B tiles -------------------
constexpr int PK_ROWS[16] = {25,40,30,51,30,51,40,71,71,25,30,20,20,512,512,32};
constexpr int PK_CA[16]   = {40,40,40,55,81,132,122,203,203,142,167,105,182,504,504,512};
constexpr int PK_CB[16]   = {25,40,30,51,30,51,40,71,71,25,30,20,20,0,0,0};
constexpr int PK_KT[16]   = {3,3,3,4,4,6,6,9,9,6,7,4,7,16,16,16};
constexpr int PK_NT[16]   = {2,3,2,4,2,4,3,5,5,2,2,2,2,32,32,2};
constexpr int PK_OFF[16]  = {0,3072,7680,10752,18944,23040,35328,44544,67584,
                             90624,96768,103936,108032,115200,377344,639488};
constexpr int PK_SA[16]   = {1,7,13,19,25,31,37,43,49,55,61,67,73,105,109,113};
constexpr int PK_SB[16]   = {3,9,15,21,27,33,39,45,51,57,63,69,75,-1,-1,-1};

__global__ __launch_bounds__(256)
void pack_kernel(Ptrs p, __bf16* __restrict__ arena) {
  const int mtx = blockIdx.x;
  const int rows = PK_ROWS[mtx], ca = PK_CA[mtx], cbn = PK_CB[mtx];
  const int kT = PK_KT[mtx];
  const float* A  = p.q[PK_SA[mtx]];
  const float* Bw = (PK_SB[mtx] >= 0) ? p.q[PK_SB[mtx]] : nullptr;
  __bf16* dst = arena + PK_OFF[mtx];
  const int total = kT * PK_NT[mtx] * 512;
  for (int e = threadIdx.x; e < total; e += 256) {
    int tile = e >> 9, w = e & 511;
    int kt = tile % kT;
    int nt = tile / kT;
    int ln = w >> 4, j = w & 15;
    int n = nt * 16 + (ln & 15);
    int k = kt * 32 + (ln >> 4) * 16 + j;
    float v = 0.f;
    if (n < rows) {
      if (k < ca)                               v = A[(size_t)n * ca + k];
      else if (Bw != nullptr && (k - ca) < cbn) v = Bw[(size_t)n * cbn + (k - ca)];
    }
    dst[e] = f2bf(v);
  }
}

// ----- persistent recurrence kernel: 1 workgroup per 16-batch tile ----------
__global__ __launch_bounds__(NTHR)
void brain_kernel(Ptrs p, float* __restrict__ out, float* __restrict__ ws,
                  const __bf16* __restrict__ arena) {
  __shared__ __attribute__((aligned(64))) float  cb  [16][CBW]; // f32 state
  __shared__ __attribute__((aligned(64))) float  inpb[16][40];  // input proj
  __shared__ __attribute__((aligned(64))) __bf16 ab  [16][ABW]; // bf16 state
  __shared__ __attribute__((aligned(64))) __bf16 xinb[16][XS];  // bf16 concat
  __shared__ __attribute__((aligned(64))) float  tgt [16][TGW];
  __shared__ float mv[16][2];

  const int tid = threadIdx.x, lane = tid & 31, wave = tid >> 5;
  const int b0 = blockIdx.x * 16;
  const float* __restrict__ xg = p.q[0];
  float* th = ws + WS_THAL + (size_t)blockIdx.x * (16 * 512);

  for (int e = tid; e < 16 * CBW; e += NTHR) ((float*)cb)[e]    = 0.f;
  for (int e = tid; e < 16 * ABW; e += NTHR) ((__bf16*)ab)[e]   = f2bf(0.f);
  for (int e = tid; e < 16 * XS;  e += NTHR) ((__bf16*)xinb)[e] = f2bf(0.f);
  __syncthreads();

  for (int t = 0; t < Tm_; ++t) {
    const float m01  = (t > 0) ? 0.1f  : 0.0f;
    const float m005 = (t > 0) ? 0.05f : 0.0f;

    // inp = input_proj @ x_t
    {
      const float* W = p.q[79]; const float* bb = p.q[80];
      for (int e = tid; e < 16 * 40; e += NTHR) {
        int row = e / 40, o = e - row * 40;
        const float* xr = xg + (((size_t)(b0 + row)) * Tm_ + t) * IN_;
        float s = bb[o];
#pragma unroll 8
        for (int k = 0; k < IN_; ++k) s += W[o * IN_ + k] * xr[k];
        inpb[row][o] = s;
      }
    }
    __syncthreads();

    // brainstem (r0): inp + m*0.1*axon(insula_to_brainstem, insula)
    seg_inp(xinb, inpb, 40, tid);
    seg(xinb, cb, 40, CO_[0], 25, tid);
    __syncthreads();
    axon_add_x(xinb, 0, 40, cb, CO_[11], 20,
               p.q[93], p.q[94], p.q[95], p.q[96], m01, tid);
    __syncthreads();
    region_stage(0, p, cb, ab, xinb, tgt, mv, arena, tid, lane, wave);

    // v1 (r1): inp + m*0.1*axon(hippo_to_v1, hippo)
    seg_inp(xinb, inpb, 40, tid);
    seg(xinb, cb, 40, CO_[1], 40, tid);
    __syncthreads();
    axon_add_x(xinb, 0, 40, cb, CO_[5], 51,
               p.q[81], p.q[82], p.q[83], p.q[84], m01, tid);
    __syncthreads();
    region_stage(1, p, cb, ab, xinb, tgt, mv, arena, tid, lane, wave);

    // v2 (r2)
    seg(xinb, cb, 0, CO_[1], 40, tid);
    seg(xinb, cb, 40, CO_[2], 30, tid);
    __syncthreads();
    region_stage(2, p, cb, ab, xinb, tgt, mv, arena, tid, lane, wave);

    // sensory (r3): [v2, brainstem]
    seg(xinb, cb, 0, CO_[2], 30, tid);
    seg(xinb, cb, 30, CO_[0], 25, tid);
    seg(xinb, cb, 55, CO_[3], 51, tid);
    __syncthreads();
    region_stage(3, p, cb, ab, xinb, tgt, mv, arena, tid, lane, wave);

    // amygdala (r4): [sen] + self
    seg(xinb, cb, 0, CO_[3], 51, tid);
    seg(xinb, cb, 51, CO_[4], 30, tid);
    seg(xinb, cb, 81, CO_[4], 30, tid);
    __syncthreads();
    region_stage(4, p, cb, ab, xinb, tgt, mv, arena, tid, lane, wave);

    // hippo (r5): [sen, amy] + self
    seg(xinb, cb, 0, CO_[3], 51, tid);
    seg(xinb, cb, 51, CO_[4], 30, tid);
    seg(xinb, cb, 81, CO_[5], 51, tid);
    seg(xinb, cb, 132, CO_[5], 51, tid);
    __syncthreads();
    region_stage(5, p, cb, ab, xinb, tgt, mv, arena, tid, lane, wave);

    // left_pf (r7): pf_in + m*0.05*axon(acc_to_left, acc_prev), self
    seg(xinb, cb, 0, CO_[3], 51, tid);
    seg(xinb, cb, 51, CO_[5], 51, tid);
    seg(xinb, cb, 102, CO_[4], 30, tid);
    seg(xinb, cb, 132, CO_[7], 71, tid);
    seg(xinb, cb, 203, CO_[7], 71, tid);
    __syncthreads();
    axon_add_x(xinb, 0, 132, cb, CO_[10], 30,
               p.q[97], p.q[98], p.q[99], p.q[100], m005, tid);
    __syncthreads();
    region_stage(7, p, cb, ab, xinb, tgt, mv, arena, tid, lane, wave);
    // lp += 0.05 * axon(amygdala_to_pf, amy)
    axon_add_cb(cb, ab, CO_[7], 71, CO_[4], 30,
                p.q[85], p.q[86], p.q[87], p.q[88], 0.05f, tid);
    __syncthreads();

    // right_pf (r8)
    seg(xinb, cb, 0, CO_[3], 51, tid);
    seg(xinb, cb, 51, CO_[5], 51, tid);
    seg(xinb, cb, 102, CO_[4], 30, tid);
    seg(xinb, cb, 132, CO_[8], 71, tid);
    seg(xinb, cb, 203, CO_[8], 71, tid);
    __syncthreads();
    axon_add_x(xinb, 0, 132, cb, CO_[10], 30,
               p.q[101], p.q[102], p.q[103], p.q[104], m005, tid);
    __syncthreads();
    region_stage(8, p, cb, ab, xinb, tgt, mv, arena, tid, lane, wave);

    // callosum (r9): [lp, rp]
    seg(xinb, cb, 0, CO_[7], 71, tid);
    seg(xinb, cb, 71, CO_[8], 71, tid);
    seg(xinb, cb, 142, CO_[9], 25, tid);
    __syncthreads();
    region_stage(9, p, cb, ab, xinb, tgt, mv, arena, tid, lane, wave);

    // acc (r10): [lp, rp, cal]
    seg(xinb, cb, 0, CO_[7], 71, tid);
    seg(xinb, cb, 71, CO_[8], 71, tid);
    seg(xinb, cb, 142, CO_[9], 25, tid);
    seg(xinb, cb, 167, CO_[10], 30, tid);
    __syncthreads();
    region_stage(10, p, cb, ab, xinb, tgt, mv, arena, tid, lane, wave);

    // insula (r11): [bs, amy, acc] + self
    seg(xinb, cb, 0, CO_[0], 25, tid);
    seg(xinb, cb, 25, CO_[4], 30, tid);
    seg(xinb, cb, 55, CO_[10], 30, tid);
    seg(xinb, cb, 85, CO_[11], 20, tid);
    seg(xinb, cb, 105, CO_[11], 20, tid);
    __syncthreads();
    region_stage(11, p, cb, ab, xinb, tgt, mv, arena, tid, lane, wave);

    // cerebellum (r6): [sen, lp]
    seg(xinb, cb, 0, CO_[3], 51, tid);
    seg(xinb, cb, 51, CO_[7], 71, tid);
    seg(xinb, cb, 122, CO_[6], 40, tid);
    __syncthreads();
    region_stage(6, p, cb, ab, xinb, tgt, mv, arena, tid, lane, wave);

    // motor (r12): [lp, rp, cer]
    seg(xinb, cb, 0, CO_[7], 71, tid);
    seg(xinb, cb, 71, CO_[8], 71, tid);
    seg(xinb, cb, 142, CO_[6], 40, tid);
    seg(xinb, cb, 182, CO_[12], 20, tid);
    __syncthreads();
    region_stage(12, p, cb, ab, xinb, tgt, mv, arena, tid, lane, wave);
    // mot += 0.1 * axon(cerebellum_to_motor, cer)
    axon_add_cb(cb, ab, CO_[12], 20, CO_[6], 40,
                p.q[89], p.q[90], p.q[91], p.q[92], 0.1f, tid);
    __syncthreads();

    // thalamus: gate = sigmoid(ln(W1 @ allr + b1)); out = gate * (W2 @ allr + b2)
    {
      const float* tb1 = p.q[106];
      const float* tg  = p.q[107];
      const float* tbt = p.q[108];
      const float* tb2 = p.q[110];
      for (int nt = wave; nt < 32; nt += NWAVES) {
        v8f acc = gemm_bf16A(&ab[0][0], ABW, 16, arena + ATH1, nt, lane);
        int col = nt * 16 + (lane & 15);
        int rbase = (lane >> 4) * 8;
#pragma unroll
        for (int i = 0; i < 8; ++i)
          th[(size_t)(rbase + i) * 512 + col] = acc[i] + tb1[col];
      }
      __threadfence();
      __syncthreads();
      if (tid < 16) {
        float mu = 0.f;
        for (int c = 0; c < 512; ++c) mu += th[(size_t)tid * 512 + c];
        mu *= (1.f / 512.f);
        float va = 0.f;
        for (int c = 0; c < 512; ++c) {
          float d = th[(size_t)tid * 512 + c] - mu; va += d * d;
        }
        va *= (1.f / 512.f);
        mv[tid][0] = mu;
        mv[tid][1] = rsqrtf(va + 1e-5f);
      }
      __syncthreads();
      for (int e = tid; e < 16 * 512; e += NTHR) {
        int row = e >> 9, c = e & 511;
        float g = (th[(size_t)row * 512 + c] - mv[row][0]) * mv[row][1] * tg[c] + tbt[c];
        th[(size_t)row * 512 + c] = 1.f / (1.f + __expf(-g));
      }
      __threadfence();
      __syncthreads();
      for (int nt = wave; nt < 32; nt += NWAVES) {
        v8f acc = gemm_bf16A(&ab[0][0], ABW, 16, arena + ATH2, nt, lane);
        int col = nt * 16 + (lane & 15);
        int rbase = (lane >> 4) * 8;
#pragma unroll
        for (int i = 0; i < 8; ++i) {
          int row = rbase + i;
          out[OUT_HID + (((size_t)(b0 + row)) * Tm_ + t) * Hm_ + col] =
              th[(size_t)row * 512 + col] * (acc[i] + tb2[col]);
        }
      }
    }
    __syncthreads();

    // traces (post-axon state, reference ORDER)
    for (int r = 0; r < 13; ++r) {
      int h = RH_[r];
      for (int e = tid; e < 16 * h; e += NTHR) {
        int row = e / h, c = e - row * h;
        out[TROFF_[r] + (((size_t)(b0 + row)) * Tm_ + t) * h + c] =
            cb[row][CO_[r] + c];
      }
    }
    __syncthreads();
  }
}

// ----- fused LayerNorm + predictions (WMMA) + anomaly head ------------------
__global__ __launch_bounds__(128)
void predanom_kernel(Ptrs p, float* __restrict__ out,
                     const __bf16* __restrict__ arena) {
  __shared__ __attribute__((aligned(64))) float  hb[16][512];
  __shared__ __attribute__((aligned(64))) __bf16 lb[16][512];
  __shared__ float mv[16][2];
  const int tid = threadIdx.x, lane = tid & 31, wave = tid >> 5;
  const size_t r0 = (size_t)blockIdx.x * 16;

  for (int e = tid; e < 16 * 512; e += 128) {
    int row = e >> 9, c = e & 511;
    hb[row][c] = out[OUT_HID + (r0 + row) * 512 + c];
  }
  __syncthreads();
  if (tid < 16) {
    float mu = 0.f;
    for (int c = 0; c < 512; ++c) mu += hb[tid][c];
    mu *= (1.f / 512.f);
    float va = 0.f;
    for (int c = 0; c < 512; ++c) { float d = hb[tid][c] - mu; va += d * d; }
    va *= (1.f / 512.f);
    mv[tid][0] = mu;
    mv[tid][1] = rsqrtf(va + 1e-5f);
  }
  __syncthreads();
  const float* ng = p.q[111]; const float* nb = p.q[112];
  for (int e = tid; e < 16 * 512; e += 128) {
    int row = e >> 9, c = e & 511;
    lb[row][c] = f2bf((hb[row][c] - mv[row][0]) * mv[row][1] * ng[c] + nb[c]);
  }
  __syncthreads();
  // anomaly = sigmoid(anom_lin(anom_ln(hidden)))
  if (tid < 16) {
    const float* ag = p.q[115]; const float* ab2 = p.q[116];
    const float* aw = p.q[117];
    float s  = p.q[118][0];
    float mu = mv[tid][0], rs = mv[tid][1];
    for (int c = 0; c < 512; ++c)
      s += aw[c] * ((hb[tid][c] - mu) * rs * ag[c] + ab2[c]);
    out[OUT_ANOM + r0 + tid] = 1.f / (1.f + __expf(-s));
  }
  // predictions = next_lin(next_ln(hidden[:, :-1]))  — WMMA 16x32x512
  if (wave < 2) {
    int nt = wave;
    v8f acc = gemm_bf16A(&lb[0][0], 512, 16, arena + ANEXT, nt, lane);
    int col = nt * 16 + (lane & 15);
    int rbase = (lane >> 4) * 8;
    float bia = p.q[114][col];
#pragma unroll
    for (int i = 0; i < 8; ++i) {
      size_t gr = r0 + rbase + i;
      int tt = (int)(gr & 255);
      if (tt < 255) {
        size_t bb = gr >> 8;
        out[OUT_PRED + (bb * 255 + tt) * 32 + col] = acc[i] + bia;
      }
    }
  }
}

// ----- surprise -------------------------------------------------------------
__global__ __launch_bounds__(256)
void surprise_kernel(const float* __restrict__ x, float* __restrict__ out) {
  int i = blockIdx.x * 256 + threadIdx.x;
  if (i >= 128 * 255) return;
  int b = i / 255, t = i - b * 255;
  const float* pr = out + OUT_PRED + (size_t)i * 32;
  const float* xr = x + (((size_t)b) * 256 + (t + 1)) * 32;
  float s = 0.f;
#pragma unroll 8
  for (int k = 0; k < 32; ++k) { float d = pr[k] - xr[k]; s += d * d; }
  out[OUT_SUR + i] = s * (1.f / 32.f);
}

// ----- props head -----------------------------------------------------------
__global__ __launch_bounds__(256)
void props_kernel(Ptrs p, float* __restrict__ out, float* __restrict__ ws) {
  float* lnp = ws + WS_PROP;        // [128][512]
  float* gbf = lnp + 128 * 512;     // [128][256]
  const int tid = threadIdx.x;
  const float* pg = p.q[119]; const float* pb = p.q[120];
  if (tid < 128) {
    const float* hr = out + OUT_HID + (((size_t)tid) * 256 + 255) * 512;
    float mu = 0.f;
    for (int c = 0; c < 512; ++c) mu += hr[c];
    mu *= (1.f / 512.f);
    float va = 0.f;
    for (int c = 0; c < 512; ++c) { float d = hr[c] - mu; va += d * d; }
    va *= (1.f / 512.f);
    float rs = rsqrtf(va + 1e-5f);
    for (int c = 0; c < 512; ++c)
      lnp[(size_t)tid * 512 + c] = (hr[c] - mu) * rs * pg[c] + pb[c];
  }
  __threadfence();
  __syncthreads();
  const float* W1 = p.q[121]; const float* b1 = p.q[122];
  for (int e = tid; e < 128 * 256; e += 256) {
    int row = e >> 8, o = e & 255;
    float s = b1[o];
    const float* wr = W1 + (size_t)o * 512;
    const float* lr = lnp + (size_t)row * 512;
    for (int c = 0; c < 512; ++c) s += wr[c] * lr[c];
    gbf[e] = 0.5f * s * (1.f + erff(s * 0.70710678f));   // exact GELU
  }
  __threadfence();
  __syncthreads();
  const float* W2 = p.q[123]; const float* b2 = p.q[124];
  for (int e = tid; e < 128 * 8; e += 256) {
    int row = e >> 3, o = e & 7;
    float s = b2[o];
    const float* wr = W2 + (size_t)o * 256;
    const float* gr = gbf + (size_t)row * 256;
    for (int c = 0; c < 256; ++c) s += wr[c] * gr[c];
    out[OUT_PROPS + e] = s;
  }
}

// ----- host ------------------------------------------------------------------
extern "C" void kernel_launch(void* const* d_in, const int* in_sizes, int n_in,
                              void* d_out, int out_size, void* d_ws, size_t ws_size,
                              hipStream_t stream) {
  (void)in_sizes; (void)out_size; (void)ws_size;
  Ptrs pp;
  for (int i = 0; i < 128; ++i) pp.q[i] = nullptr;
  int n = n_in < 128 ? n_in : 128;
  for (int i = 0; i < n; ++i) pp.q[i] = (const float*)d_in[i];

  float*  out   = (float*)d_out;
  float*  ws    = (float*)d_ws;
  __bf16* arena = (__bf16*)d_ws;

  pack_kernel<<<dim3(16), dim3(256), 0, stream>>>(pp, arena);
  brain_kernel<<<dim3(8), dim3(NTHR), 0, stream>>>(pp, out, ws,
                                                   (const __bf16*)arena);
  predanom_kernel<<<dim3(2048), dim3(128), 0, stream>>>(pp, out,
                                                        (const __bf16*)arena);
  surprise_kernel<<<dim3(128), dim3(256), 0, stream>>>(pp.q[0], out);
  props_kernel<<<dim3(1), dim3(256), 0, stream>>>(pp, out, ws);
}
